// BitAttention_360777253468
// MI455X (gfx1250) — matrix operand order repaired and checked by reading, not compiled, not run on hardware
//
#include <hip/hip_runtime.h>

// ---------------------------------------------------------------- types
typedef __bf16 bf16;
typedef bf16  v16bf __attribute__((ext_vector_type(16)));
typedef float v8f   __attribute__((ext_vector_type(8)));
typedef unsigned short u16;
typedef unsigned int   u32;
typedef unsigned int v4u __attribute__((ext_vector_type(4)));
typedef int          v8i __attribute__((ext_vector_type(8)));
typedef int          v4i __attribute__((ext_vector_type(4)));

union FragBF {
  v16bf v;
  uint4 q[2];
  u16   s[16];
};

// round-to-nearest-even float -> bf16 (as u16)
__device__ __forceinline__ u16 f2bf(float f) {
  union { float f; u32 u; } v; v.f = f;
  u32 r = v.u + 0x7fffu + ((v.u >> 16) & 1u);
  return (u16)(r >> 16);
}

// ---------------------------------------------------------------- WMMA fragment loaders
// CDNA5 16-bit A-matrix 16x32 (MxK), wave32:
//   lane = 16*hi + r : row M=r ; VGPR0..3 hold K = 8*hi + [0..8) ; VGPR4..7 hold K = 16 + 8*hi + [0..8)
__device__ __forceinline__ v16bf load_a_frag(const u16* base /*(row0,k0)*/, int lda, int r, int hi) {
  FragBF f;
  const u16* p = base + (size_t)r * lda + 8 * hi;
  f.q[0] = *(const uint4*)(p);        // K chunk [8hi, 8hi+8)
  f.q[1] = *(const uint4*)(p + 16);   // K chunk [16+8hi, 16+8hi+8)
  return f.v;
}
// CDNA5 16-bit B-matrix 32x16 (KxN): lane (r,hi): column N=r, VGPR0..3 hold K = 8*hi + [0..8) ; VGPR4..7 hold K = 16 + 8*hi + [0..8)
// Memory holds W[n][k] row-major (ld = ldb), i.e. B[k][n] = W[n][k] -> contiguous in k.
__device__ __forceinline__ v16bf load_b_frag(const u16* base /*(n0 row,k0 col)*/, int ldb, int r, int hi) {
  FragBF f;
  const u16* p = base + (size_t)r * ldb + 8 * hi;
  f.q[0] = *(const uint4*)(p);
  f.q[1] = *(const uint4*)(p + 16);
  return f.v;
}

__device__ __forceinline__ v8f wmma_bf16(v16bf a, v16bf b, v8f c) {
  return __builtin_amdgcn_wmma_f32_16x16x32_bf16(false, a, false, b, (short)0, c, false, false);
}

// ---------------------------------------------------------------- TDM: stage a 256(n) x 64(k) bf16
// weight tile (rows of W[n][k], ld=2048) into LDS, row-major contiguous (ld=64 elems).
// D# built in SGPRs (all inputs are block-uniform); issued as raw tensor_load_to_lds
// (inline asm = portable across ROCm7.2 / clang-23 builtin-arity split). Tracked by TENSORcnt.
__device__ __forceinline__ void tdm_load_w_tile(const u16* gsrc, u32 lds_byte_off) {
  unsigned long long ga = (unsigned long long)(size_t)gsrc;
  v4u g0;
  g0.x = 1u;                                              // count=1 user descriptor
  g0.y = lds_byte_off;                                    // lds_addr (bytes)
  g0.z = (u32)ga;                                         // global_addr[31:0]
  g0.w = ((u32)(ga >> 32) & 0x01FFFFFFu) | 0x80000000u;   // global_addr[56:32] | type=2
  v8i g1;
  g1[0] = 0x00010000;                                     // data_size=1 (2-byte elems)
  g1[1] = (int)(2048u << 16);                             // tensor_dim0 = 2048 (bits 79:48)
  g1[2] = (int)(2048u << 16);                             // tensor_dim1 = 2048 (bits 111:80)
  g1[3] = (int)(64u << 16);                               // tile_dim0 = 64 (bits 127:112)
  g1[4] = 256;                                            // tile_dim1 = 256 ; tile_dim2 = 0
  g1[5] = 2048;                                           // tensor_dim0_stride[31:0] = 2048
  g1[6] = (int)(2048u << 16);                             // tensor_dim1_stride[15:0] = 2048
  g1[7] = 0;                                              // tensor_dim1_stride[47:16] = 0
  v4i gz = {0, 0, 0, 0};                                  // groups 2/3: tile_dim3=0 (2D tile)
  asm volatile("tensor_load_to_lds %0, %1, %2, %3"
               :: "s"(g0), "s"(g1), "s"(gz), "s"(gz)
               : "memory");
}

// ---------------------------------------------------------------- reductions / quant / cast
#define DD 4194304   // 2048*2048
#define CHUNK 65536  // DD / 64 blocks

__global__ void absmean_partial_kernel(const float* __restrict__ w, float* __restrict__ part) {
  __shared__ float red[256];
  size_t base = (size_t)blockIdx.x * CHUNK;
  float s = 0.f;
  for (int i = threadIdx.x; i < CHUNK; i += 256) s += fabsf(w[base + i]);
  red[threadIdx.x] = s;
  __syncthreads();
  for (int st = 128; st > 0; st >>= 1) {
    if ((int)threadIdx.x < st) red[threadIdx.x] += red[threadIdx.x + st];
    __syncthreads();
  }
  if (threadIdx.x == 0) part[blockIdx.x] = red[0];
}

__global__ void finalize_scales_kernel(const float* __restrict__ part, float* __restrict__ scales) {
  int w = threadIdx.x;
  if (w < 4) {
    float s = 0.f;
    for (int i = 0; i < 64; ++i) s += part[w * 64 + i];   // fixed order -> deterministic
    float m = s * (1.0f / (float)DD);
    scales[w] = m > 1e-5f ? m : 1e-5f;
  }
}

__global__ void quantize_w_kernel(const float* __restrict__ w, const float* __restrict__ scalep,
                                  u16* __restrict__ o) {
  int i = blockIdx.x * 256 + threadIdx.x;
  float sc = *scalep;
  float v = w[i];
  float q = v > 0.f ? sc : (v < 0.f ? -sc : 0.f);
  o[i] = f2bf(q);
}

__global__ void cast_x_kernel(const float* __restrict__ x, u16* __restrict__ o) {
  int i = blockIdx.x * 256 + threadIdx.x;
  o[i] = f2bf(x[i]);
}

// ---------------------------------------------------------------- WMMA GEMM: Y[4096,2048] = A @ W^T
// block = 256 thr (8 waves); block tile 32(M) x 256(N); K-step 64.
// Weight tile (256x64) TDM-staged into double-buffered LDS; A fragments register-pipelined;
// all 8 B fragments batch-loaded from LDS before the WMMA chain (partial dscnt waits).
// mode 1: bf16 out as [b,h,t,hd]   (Q / K buffers)
// mode 2: bf16 out as [b,h,hd,t]   (V transposed)
// mode 3: f32 plain [M,N]          (final output)
__global__ __launch_bounds__(256)
void gemm_bf16_wmma_kernel(const u16* __restrict__ A, const u16* __restrict__ W,
                           int mode, u16* __restrict__ outb, float* __restrict__ outf) {
  __shared__ __align__(16) u16 bsm[2][256 * 64];   // 2 x 32 KB weight tiles

  const int lane = threadIdx.x & 31;
  const int r = lane & 15, hi = lane >> 4;
  const int wid = threadIdx.x >> 5;
  const int wm = wid >> 2, wn = wid & 3;
  const int row0 = blockIdx.y * 32 + wm * 16;
  const int col0 = blockIdx.x * 256;
  const int K = 2048, N = 2048;
  const int NK = K / 64;

  // prologue: TDM weight tile for iteration 0
  if (threadIdx.x < 32)
    tdm_load_w_tile(W + (size_t)col0 * K, (u32)(size_t)(void*)&bsm[0][0]);

  const v8f vzero = {0.f,0.f,0.f,0.f,0.f,0.f,0.f,0.f};
  v8f acc[4] = {vzero, vzero, vzero, vzero};

  // prologue: A fragments for iteration 0 (two 32-K chunks of the 64-K step)
  FragBF a[2];
  a[0].v = load_a_frag(A + (size_t)row0 * K + 0,  K, r, hi);
  a[1].v = load_a_frag(A + (size_t)row0 * K + 32, K, r, hi);

  for (int it = 0; it < NK; ++it) {
    const int k0 = it * 64;
    if (threadIdx.x < 32) {
      if (it + 1 < NK) {
        tdm_load_w_tile(W + (size_t)col0 * K + (k0 + 64),
                        (u32)(size_t)(void*)&bsm[(it + 1) & 1][0]);
        asm volatile("s_wait_tensorcnt 0x1" ::: "memory");   // current tile resident
      } else {
        asm volatile("s_wait_tensorcnt 0x0" ::: "memory");
      }
    }
    __syncthreads();

    // software-pipeline: issue next iteration's A loads before consuming current
    FragBF an[2];
    if (it + 1 < NK) {
      an[0].v = load_a_frag(A + (size_t)row0 * K + k0 + 64, K, r, hi);
      an[1].v = load_a_frag(A + (size_t)row0 * K + k0 + 96, K, r, hi);
    }

    // batch all B fragments from LDS, then run the WMMA chain
    const u16* bs = &bsm[it & 1][0];
    FragBF bfr[8];
#pragma unroll
    for (int kc = 0; kc < 2; ++kc)
#pragma unroll
      for (int nt = 0; nt < 4; ++nt)
        bfr[kc * 4 + nt].v =
            load_b_frag(bs + (size_t)(wn * 64 + nt * 16) * 64 + kc * 32, 64, r, hi);
#pragma unroll
    for (int kc = 0; kc < 2; ++kc)
#pragma unroll
      for (int nt = 0; nt < 4; ++nt)
        acc[nt] = wmma_bf16(a[kc].v, bfr[kc * 4 + nt].v, acc[nt]);

    __syncthreads();                 // tile consumed before TDM overwrites this buffer
    if (it + 1 < NK) { a[0] = an[0]; a[1] = an[1]; }
  }

#pragma unroll
  for (int nt = 0; nt < 4; ++nt) {
#pragma unroll
    for (int v = 0; v < 8; ++v) {
      const int m = v + 8 * hi;               // C/D layout: m = vgpr + 8*hi, n = lane&15
      const int row = row0 + m;
      const int col = col0 + wn * 64 + nt * 16 + r;
      const float val = acc[nt][v];
      if (mode == 3) {
        outf[(size_t)row * N + col] = val;
      } else {
        const int b = row >> 11, t = row & 2047;   // T = 2048
        const int h = col >> 7,  d = col & 127;    // hd = 128
        if (mode == 1)
          outb[(((size_t)(b * 16 + h) * 2048 + t) << 7) + d] = f2bf(val);
        else
          outb[((size_t)(b * 16 + h) * 128 + d) * 2048 + t] = f2bf(val);
      }
    }
  }
}

// ---------------------------------------------------------------- FlashAttention (causal, online softmax)
// 128 thr = 4 waves/block; one wave owns 16 query rows of one (b,h).
// Q,K: [bh][2048][128] bf16 ; Vt: [bh][128][2048] bf16 ; Y: [b*2048+t][2048] bf16.
// Per kv iteration: issue ALL K and V fragment loads first (V latency hides behind the
// S WMMAs + softmax VALU work), then S WMMAs, softmax, P staging, O WMMAs.
__global__ __launch_bounds__(128)
void flash_attn_wmma_kernel(const u16* __restrict__ Qb, const u16* __restrict__ Kb,
                            const u16* __restrict__ Vt, u16* __restrict__ Y) {
  __shared__ __align__(16) u16 plds[4][16 * 32];   // per-wave P staging (16q x 32kv bf16)

  const int lane = threadIdx.x & 31;
  const int r = lane & 15, hi = lane >> 4;
  const int wid = threadIdx.x >> 5;
  const int gw = blockIdx.x * 4 + wid;
  const int bh = gw >> 7;          // 128 q-blocks per (b,h)
  const int q0 = (gw & 127) * 16;

  const u16* Qh = Qb + (size_t)bh * 2048 * 128;
  const u16* Kh = Kb + (size_t)bh * 2048 * 128;
  const u16* Vh = Vt + (size_t)bh * 128 * 2048;

  // Q fragments resident for all kv chunks (4 x K32 chunks covering hd=128)
  v16bf qf[4];
#pragma unroll
  for (int c = 0; c < 4; ++c)
    qf[c] = load_a_frag(Qh + (size_t)q0 * 128 + c * 32, 128, r, hi);

  const v8f vzero = {0.f,0.f,0.f,0.f,0.f,0.f,0.f,0.f};
  v8f o[8] = {vzero, vzero, vzero, vzero, vzero, vzero, vzero, vzero};
  float mrow[8], lrow[8];
#pragma unroll
  for (int v = 0; v < 8; ++v) { mrow[v] = -1e30f; lrow[v] = 0.f; }

  const float scale = 0.08838834764831845f;   // 1/sqrt(128)

  for (int kv0 = 0; kv0 <= q0; kv0 += 32) {
    // ---- batch-issue all global fragment loads for this kv chunk
    FragBF kb0[4], kb1[4], vb[8];
#pragma unroll
    for (int c = 0; c < 4; ++c) {
      kb0[c].v = load_b_frag(Kh + (size_t)(kv0)      * 128 + c * 32, 128, r, hi);
      kb1[c].v = load_b_frag(Kh + (size_t)(kv0 + 16) * 128 + c * 32, 128, r, hi);
    }
#pragma unroll
    for (int nt = 0; nt < 8; ++nt)
      vb[nt].v = load_b_frag(Vh + (size_t)(nt * 16) * 2048 + kv0, 2048, r, hi);

    // ---- S = Q K^T  (two 16-col kv tiles, K-loop over hd)
    v8f s0 = vzero, s1 = vzero;
#pragma unroll
    for (int c = 0; c < 4; ++c) s0 = wmma_bf16(qf[c], kb0[c].v, s0);
#pragma unroll
    for (int c = 0; c < 4; ++c) s1 = wmma_bf16(qf[c], kb1[c].v, s1);

    // ---- online softmax on C-layout tiles; write P (bf16) to per-wave LDS
#pragma unroll
    for (int v = 0; v < 8; ++v) {
      const int m = v + 8 * hi;
      const int qrow = q0 + m;
      float e0 = s0[v] * scale;
      float e1 = s1[v] * scale;
      if (kv0 + r > qrow)      e0 = -1e30f;   // causal mask
      if (kv0 + 16 + r > qrow) e1 = -1e30f;
      float mx = fmaxf(e0, e1);
      mx = fmaxf(mx, __shfl_xor(mx, 1, 32));
      mx = fmaxf(mx, __shfl_xor(mx, 2, 32));
      mx = fmaxf(mx, __shfl_xor(mx, 4, 32));
      mx = fmaxf(mx, __shfl_xor(mx, 8, 32));
      const float mnew = fmaxf(mrow[v], mx);
      const float corr = __expf(mrow[v] - mnew);
      mrow[v] = mnew;
      const float p0 = __expf(e0 - mnew);
      const float p1 = __expf(e1 - mnew);
      float ps = p0 + p1;
      ps += __shfl_xor(ps, 1, 32);
      ps += __shfl_xor(ps, 2, 32);
      ps += __shfl_xor(ps, 4, 32);
      ps += __shfl_xor(ps, 8, 32);
      lrow[v] = lrow[v] * corr + ps;
#pragma unroll
      for (int nt = 0; nt < 8; ++nt) o[nt][v] *= corr;
      plds[wid][m * 32 + r]      = f2bf(p0);
      plds[wid][m * 32 + 16 + r] = f2bf(p1);
    }

    // same-wave LDS RAW: LDS ops are in-order per wave; make the dependency explicit
    asm volatile("s_wait_dscnt 0x0" ::: "memory");

    // ---- O += P V   (A = P from LDS, B fragments already resident)
    v16bf pa = load_a_frag(&plds[wid][0], 32, r, hi);
#pragma unroll
    for (int nt = 0; nt < 8; ++nt)
      o[nt] = wmma_bf16(pa, vb[nt].v, o[nt]);
  }

  // ---- normalize + store Y[b*2048 + t][h*128 + d]
  const int b = bh >> 4, h = bh & 15;
#pragma unroll
  for (int v = 0; v < 8; ++v) {
    const int m = v + 8 * hi;
    const float inv = 1.0f / lrow[v];
    const size_t base = ((size_t)(b * 2048 + q0 + m)) * 2048 + h * 128;
#pragma unroll
    for (int nt = 0; nt < 8; ++nt)
      Y[base + nt * 16 + r] = f2bf(o[nt][v] * inv);
  }
}

// ---------------------------------------------------------------- launch
extern "C" void kernel_launch(void* const* d_in, const int* in_sizes, int n_in,
                              void* d_out, int out_size, void* d_ws, size_t ws_size,
                              hipStream_t stream) {
  const float* x  = (const float*)d_in[0];
  const float* wq = (const float*)d_in[1];
  const float* wk = (const float*)d_in[2];
  const float* wv = (const float*)d_in[3];
  const float* wo = (const float*)d_in[4];
  float* out = (float*)d_out;

  char* ws = (char*)d_ws;
  float* part   = (float*)(ws);              // 4*64 floats
  float* scales = (float*)(ws + 1024);       // 4 floats
  size_t off = 4096;
  const size_t MAT = (size_t)4096 * 2048;    // elements of a [4096,2048] bf16 buffer
  u16* xbf  = (u16*)(ws + off); off += MAT * 2;
  u16* wbf0 = (u16*)(ws + off); off += (size_t)DD * 2;
  u16* wbf1 = (u16*)(ws + off); off += (size_t)DD * 2;
  u16* wbf2 = (u16*)(ws + off); off += (size_t)DD * 2;
  u16* wbf3 = (u16*)(ws + off); off += (size_t)DD * 2;
  u16* qb   = (u16*)(ws + off); off += MAT * 2;
  u16* kb   = (u16*)(ws + off); off += MAT * 2;
  u16* vtb  = (u16*)(ws + off); off += MAT * 2;
  u16* yatt = (u16*)(ws + off); off += MAT * 2;

  const float* wsrc[4] = {wq, wk, wv, wo};
  u16* wdst[4] = {wbf0, wbf1, wbf2, wbf3};

  // 1) deterministic mean(|w|) per weight
  for (int i = 0; i < 4; ++i)
    absmean_partial_kernel<<<64, 256, 0, stream>>>(wsrc[i], part + 64 * i);
  finalize_scales_kernel<<<1, 4, 0, stream>>>(part, scales);

  // 2) quantize weights to +/- scale (bf16); cast x to bf16
  for (int i = 0; i < 4; ++i)
    quantize_w_kernel<<<DD / 256, 256, 0, stream>>>(wsrc[i], scales + i, wdst[i]);
  cast_x_kernel<<<(2 * DD) / 256, 256, 0, stream>>>(x, xbf);

  // 3) projections; grid: x = N/256, y = M/32
  dim3 ggrid(8, 128);
  gemm_bf16_wmma_kernel<<<ggrid, 256, 0, stream>>>(xbf, wbf0, 1, qb,  nullptr); // Q
  gemm_bf16_wmma_kernel<<<ggrid, 256, 0, stream>>>(xbf, wbf1, 1, kb,  nullptr); // K
  gemm_bf16_wmma_kernel<<<ggrid, 256, 0, stream>>>(xbf, wbf2, 2, vtb, nullptr); // V^T

  // 4) causal flash attention: 32 bh * 128 qblocks = 4096 waves / 4 per block
  flash_attn_wmma_kernel<<<1024, 128, 0, stream>>>(qb, kb, vtb, yatt);

  // 5) output projection -> f32 d_out
  gemm_bf16_wmma_kernel<<<ggrid, 256, 0, stream>>>(yatt, wbf3, 3, nullptr, out);

  (void)in_sizes; (void)n_in; (void)out_size; (void)ws_size;
}
